// HCA_53635551592624
// MI455X (gfx1250) — matrix operand order repaired and checked by reading, not compiled, not run on hardware
//
#include <hip/hip_runtime.h>
#include <cstdint>
#include <cstddef>

// ---------------------------------------------------------------------------
// Types for CDNA5 WMMA (wave32): v_wmma_f32_16x16x32_bf16
// ---------------------------------------------------------------------------
typedef __attribute__((ext_vector_type(16))) __bf16 v16bf;
typedef __attribute__((ext_vector_type(8)))  float  v8f;
typedef __attribute__((ext_vector_type(4)))  unsigned int u32x4;
typedef __attribute__((ext_vector_type(4)))  __bf16 bf16x4;

union FragU { v16bf v; u32x4 q[2]; };

// A-fragment: 16x32 bf16 slice starting at p (p already includes the per-lane
// row*ldk + ((lane>>4)<<4) offset). The per-lane data is
// K = {kb..kb+15} contiguous (matches B): two 16B chunks.
__device__ __forceinline__ v16bf ld_a(const __bf16* p) {
  FragU u;
  u.q[0] = *(const u32x4*)(p);
  u.q[1] = *(const u32x4*)(p + 8);
  return u.v;
}

// B-fragment: 32x16 bf16; Wt stored [col][k] so the 16 per-lane K values are
// contiguous (lanes 0-15: K=0..15, lanes 16-31: K=16..31): two 16B chunks.
__device__ __forceinline__ v16bf ld_b(const __bf16* p) {
  FragU u;
  u.q[0] = *(const u32x4*)(p);
  u.q[1] = *(const u32x4*)(p + 8);
  return u.v;
}

#define WMMA_BF16(a, b, c) \
  __builtin_amdgcn_wmma_f32_16x16x32_bf16(false, (a), false, (b), (short)0, (c), false, false)

__device__ __forceinline__ float softplusf(float v) {
  return fmaxf(v, 0.0f) + log1pf(expf(-fabsf(v)));
}

// ---------------------------------------------------------------------------
// Weight prep: transpose + bf16 convert.  wt[c*K + k] = bf16(w[k*Ncols + c])
// ---------------------------------------------------------------------------
__global__ void k_transpose_bf16(const float* __restrict__ w, unsigned short* wt_,
                                 int K, int Ncols) {
  __bf16* wt = (__bf16*)wt_;
  size_t idx = (size_t)blockIdx.x * blockDim.x + threadIdx.x;
  size_t total = (size_t)K * Ncols;
  if (idx >= total) return;
  int c = (int)(idx / K);
  int k = (int)(idx % K);
  wt[idx] = (__bf16)w[(size_t)k * Ncols + c];
}

__global__ void k_transpose_f32(const float* __restrict__ w, float* __restrict__ wt,
                                int K, int Ncols) {
  size_t idx = (size_t)blockIdx.x * blockDim.x + threadIdx.x;
  size_t total = (size_t)K * Ncols;
  if (idx >= total) return;
  int c = (int)(idx / K);
  int k = (int)(idx % K);
  wt[idx] = w[(size_t)k * Ncols + c];
}

// ---------------------------------------------------------------------------
// pre1: h = bf16(tanh(x @ W_pre1 + b1)), K=16 -> plain VALU, fp32 accumulate.
// ---------------------------------------------------------------------------
__global__ __launch_bounds__(256) void k_pre1(const float* __restrict__ x,
                                              const float* __restrict__ w1t, // [H][16]
                                              const float* __restrict__ b1,
                                              unsigned short* h_, int H) {
  __bf16* h = (__bf16*)h_;
  const int n = blockIdx.x;
  const float* xp = x + (size_t)n * 16;
  float xr[16];
#pragma unroll
  for (int i = 0; i < 16; ++i) xr[i] = xp[i];
  for (int j = threadIdx.x; j < H; j += blockDim.x) {
    const float* wp = w1t + (size_t)j * 16;
    float acc = b1[j];
#pragma unroll
    for (int i = 0; i < 16; ++i) acc += xr[i] * wp[i];
    h[(size_t)n * H + j] = (__bf16)tanhf(acc);
  }
}

// ---------------------------------------------------------------------------
// Main GEMM: C_bf16 = (tanh?)(A_bf16[N,K] @ Wt_bf16[Ncols,K]^T + bias)
// Block = 8 waves, block tile 256(M) x 64(N); wave tile = 32x64:
//   2 A-frags x 4 B-frags -> 8 accumulators, 8 WMMAs per 32-wide K step.
// Register double-buffering: next K-chunk's 6 fragments are loaded before the
// current chunk's WMMAs issue (loop is branch-free; final prefetch runs one
// chunk past K and is covered by a 256B pad after each buffer).
// ---------------------------------------------------------------------------
__global__ __launch_bounds__(256) void k_gemm_bf16(
    const unsigned short* A_, const unsigned short* Wt_,
    const float* __restrict__ bias, unsigned short* C_,
    int Nrows, int K, int Ncols, int apply_tanh) {
  const __bf16* A  = (const __bf16*)A_;
  const __bf16* Wt = (const __bf16*)Wt_;
  __bf16* C = (__bf16*)C_;

  const int lane = threadIdx.x & 31;
  const int wave = threadIdx.x >> 5;
  const int colTiles = Ncols >> 6;
  const int bx = blockIdx.x % colTiles;
  const int by = blockIdx.x / colTiles;
  const int row0 = by * 256 + wave * 32;
  const int col0 = bx * 64;
  if (row0 >= Nrows) return;                 // uniform per wave: EXEC stays all-1s

  const int rA0 = min(row0 + (lane & 15), Nrows - 1);
  const int rA1 = min(row0 + 16 + (lane & 15), Nrows - 1);
  const int cB  = lane & 15;

  const __bf16* a0p = A + (size_t)rA0 * K + ((lane >> 4) << 4);
  const __bf16* a1p = A + (size_t)rA1 * K + ((lane >> 4) << 4);
  const __bf16* b0p = Wt + (size_t)(col0 +  0 + cB) * K + ((lane >> 4) << 4);
  const __bf16* b1p = Wt + (size_t)(col0 + 16 + cB) * K + ((lane >> 4) << 4);
  const __bf16* b2p = Wt + (size_t)(col0 + 32 + cB) * K + ((lane >> 4) << 4);
  const __bf16* b3p = Wt + (size_t)(col0 + 48 + cB) * K + ((lane >> 4) << 4);

  v8f acc[8] = {v8f{0.f}, v8f{0.f}, v8f{0.f}, v8f{0.f},
                v8f{0.f}, v8f{0.f}, v8f{0.f}, v8f{0.f}};

  // Prologue: fragments for kc = 0
  v16bf a0 = ld_a(a0p), a1 = ld_a(a1p);
  v16bf b0 = ld_b(b0p), b1 = ld_b(b1p), b2 = ld_b(b2p), b3 = ld_b(b3p);

  for (int kc = 0; kc < K; kc += 32) {
    a0p += 32; a1p += 32; b0p += 32; b1p += 32; b2p += 32; b3p += 32;
    // Prefetch next chunk (last iteration reads <=32 elems past K: padded).
    v16bf a0n = ld_a(a0p), a1n = ld_a(a1p);
    v16bf b0n = ld_b(b0p), b1n = ld_b(b1p), b2n = ld_b(b2p), b3n = ld_b(b3p);

    acc[0] = WMMA_BF16(a0, b0, acc[0]);
    acc[1] = WMMA_BF16(a0, b1, acc[1]);
    acc[2] = WMMA_BF16(a0, b2, acc[2]);
    acc[3] = WMMA_BF16(a0, b3, acc[3]);
    acc[4] = WMMA_BF16(a1, b0, acc[4]);
    acc[5] = WMMA_BF16(a1, b1, acc[5]);
    acc[6] = WMMA_BF16(a1, b2, acc[6]);
    acc[7] = WMMA_BF16(a1, b3, acc[7]);

    a0 = a0n; a1 = a1n; b0 = b0n; b1 = b1n; b2 = b2n; b3 = b3n;
  }

  // Epilogue: lane L -> col = col0+t*16+(L&15); rows = row0+g*16+vr+8*(L>=16)
  const int rbase = row0 + ((lane >> 4) << 3);
#pragma unroll
  for (int t = 0; t < 4; ++t) {
    const int cc = col0 + t * 16 + cB;
    const float bj = bias[cc];
#pragma unroll
    for (int g = 0; g < 2; ++g) {
#pragma unroll
      for (int vr = 0; vr < 8; ++vr) {
        const int rr = rbase + g * 16 + vr;
        if (rr < Nrows) {
          float v = acc[g * 4 + t][vr] + bj;
          if (apply_tanh) v = tanhf(v);
          C[(size_t)rr * Ncols + cc] = (__bf16)v;
        }
      }
    }
  }
}

// ---------------------------------------------------------------------------
// z = 0
// ---------------------------------------------------------------------------
__global__ void k_zero_f32(float* z, size_t n4) {  // n4 = total/4
  size_t i = (size_t)blockIdx.x * blockDim.x + threadIdx.x;
  if (i < n4) *(float4*)(z + i * 4) = make_float4(0.f, 0.f, 0.f, 0.f);
}

// ---------------------------------------------------------------------------
// Edge aggregate: z[dst] += m_bf16[src]  (fp32 atomics, one block per edge)
// ---------------------------------------------------------------------------
__global__ __launch_bounds__(256) void k_aggregate(
    const unsigned short* m_, const int* __restrict__ esrc,
    const int* __restrict__ edst, float* z, int H) {
  const __bf16* m = (const __bf16*)m_;
  const int e = blockIdx.x;
  const int s = esrc[e];
  const int d = edst[e];
  const __bf16* mp = m + (size_t)s * H;
  float* zp = z + (size_t)d * H;
  for (int j = threadIdx.x * 4; j < H; j += blockDim.x * 4) {
    bf16x4 v = *(const bf16x4*)(mp + j);     // 8B coalesced gather
#pragma unroll
    for (int i = 0; i < 4; ++i) atomicAdd(zp + j + i, (float)v[i]);
  }
}

// ---------------------------------------------------------------------------
// zt = bf16(tanh(z))
// ---------------------------------------------------------------------------
__global__ void k_tanh_z(const float* __restrict__ z, unsigned short* zt_, size_t total) {
  __bf16* zt = (__bf16*)zt_;
  size_t i = (size_t)blockIdx.x * blockDim.x + threadIdx.x;
  if (i < total) zt[i] = (__bf16)tanhf(z[i]);
}

// ---------------------------------------------------------------------------
// post: out = softplus([zt | h] @ W_post + b_post)   (fused concat, F=16 cols)
// ---------------------------------------------------------------------------
__global__ __launch_bounds__(256) void k_post(
    const unsigned short* Zt_, const unsigned short* Hb_, const unsigned short* Wpt_,
    const float* __restrict__ bias, float* __restrict__ out, int Nrows, int H) {
  const __bf16* Zt  = (const __bf16*)Zt_;
  const __bf16* Hb  = (const __bf16*)Hb_;
  const __bf16* Wpt = (const __bf16*)Wpt_;   // [16][2H]

  const int lane = threadIdx.x & 31;
  const int wave = threadIdx.x >> 5;
  const int row0 = (blockIdx.x * 8 + wave) * 16;
  if (row0 >= Nrows) return;

  const int rA = min(row0 + (lane & 15), Nrows - 1);
  const int cB = lane & 15;
  const int ldb = 2 * H;

  const __bf16* azp = Zt + (size_t)rA * H + ((lane >> 4) << 4);
  const __bf16* ahp = Hb + (size_t)rA * H + ((lane >> 4) << 4);
  const __bf16* bp  = Wpt + (size_t)cB * ldb + ((lane >> 4) << 4);

  v8f acc = v8f{0.f};
  for (int kc = 0; kc < H; kc += 32) {       // z half (k in [0,H))
    acc = WMMA_BF16(ld_a(azp + kc), ld_b(bp + kc), acc);
  }
  for (int kc = 0; kc < H; kc += 32) {       // h half (k in [H,2H))
    acc = WMMA_BF16(ld_a(ahp + kc), ld_b(bp + H + kc), acc);
  }

  const float bj = bias[cB];
  const int rbase = row0 + ((lane >> 4) << 3);
#pragma unroll
  for (int vr = 0; vr < 8; ++vr) {
    const int rr = rbase + vr;
    if (rr < Nrows)
      out[(size_t)rr * 16 + cB] = softplusf(acc[vr] + bj);
  }
}

// ---------------------------------------------------------------------------
// Launch
// ---------------------------------------------------------------------------
extern "C" void kernel_launch(void* const* d_in, const int* in_sizes, int n_in,
                              void* d_out, int out_size, void* d_ws, size_t ws_size,
                              hipStream_t stream) {
  const float* x      = (const float*)d_in[0];
  const int*   esrc   = (const int*)d_in[1];
  const int*   edst   = (const int*)d_in[2];
  const float* W_pre1 = (const float*)d_in[3];
  const float* b_pre1 = (const float*)d_in[4];
  const float* W_pre2 = (const float*)d_in[5];
  const float* b_pre2 = (const float*)d_in[6];
  const float* W_mp   = (const float*)d_in[7];
  const float* b_mp   = (const float*)d_in[8];
  const float* W_post = (const float*)d_in[9];
  const float* b_post = (const float*)d_in[10];

  const int F = 16;
  const int N = in_sizes[0] / F;       // 50000
  const int E = in_sizes[1];           // 400000
  const int H = in_sizes[3] / F;       // 1024
  const size_t NH = (size_t)N * H;
  const size_t HH = (size_t)H * H;
  const size_t PAD = 256;              // bytes; covers one-chunk prefetch overrun

  // Workspace carve-out
  char* p = (char*)d_ws;
  unsigned short* bufA   = (unsigned short*)p; p += NH * 2 + PAD;       // h1 -> m -> zt
  unsigned short* bufB   = (unsigned short*)p; p += NH * 2 + PAD;       // h
  float*          z      = (float*)p;          p += NH * 4;             // fp32 accum
  unsigned short* Wt2    = (unsigned short*)p; p += HH * 2 + PAD;       // W_pre2^T bf16
  unsigned short* Wmpt   = (unsigned short*)p; p += HH * 2 + PAD;       // W_mp^T bf16
  unsigned short* Wpostt = (unsigned short*)p; p += (size_t)2 * H * F * 2 + PAD; // W_post^T
  float*          Wpre1t = (float*)p;          p += (size_t)F * H * 4;  // W_pre1^T f32

  // 1) weight prep
  k_transpose_bf16<<<(unsigned)((HH + 255) / 256), 256, 0, stream>>>(W_pre2, Wt2, H, H);
  k_transpose_bf16<<<(unsigned)((HH + 255) / 256), 256, 0, stream>>>(W_mp, Wmpt, H, H);
  k_transpose_bf16<<<(unsigned)(((size_t)2 * H * F + 255) / 256), 256, 0, stream>>>(W_post, Wpostt, 2 * H, F);
  k_transpose_f32 <<<(unsigned)(((size_t)F * H + 255) / 256), 256, 0, stream>>>(W_pre1, Wpre1t, F, H);

  // 2) h1 = tanh(x @ W1 + b1) -> bufA (bf16)
  k_pre1<<<N, 256, 0, stream>>>(x, Wpre1t, b_pre1, bufA, H);

  // 3) h = tanh(h1 @ W2 + b2) -> bufB ;  4) m = h @ Wmp + bmp -> bufA
  const int colTiles  = H / 64;
  const int rowBlocks = (N + 255) / 256;
  const unsigned gemmGrid = (unsigned)(rowBlocks * colTiles);
  k_gemm_bf16<<<gemmGrid, 256, 0, stream>>>(bufA, Wt2,  b_pre2, bufB, N, H, H, /*tanh=*/1);
  k_gemm_bf16<<<gemmGrid, 256, 0, stream>>>(bufB, Wmpt, b_mp,   bufA, N, H, H, /*tanh=*/0);

  // 5) z = 0 ; 6) z[dst] += m[src]
  k_zero_f32<<<(unsigned)((NH / 4 + 255) / 256), 256, 0, stream>>>(z, NH / 4);
  k_aggregate<<<(unsigned)E, 256, 0, stream>>>(bufA, esrc, edst, z, H);

  // 7) zt = bf16(tanh(z)) -> bufA (m is dead)
  k_tanh_z<<<(unsigned)((NH + 255) / 256), 256, 0, stream>>>(z, bufA, NH);

  // 8) out = softplus([zt|h] @ W_post + b_post)
  const int tiles = (N + 15) / 16;
  k_post<<<(unsigned)((tiles + 7) / 8), 256, 0, stream>>>(bufA, bufB, Wpostt, b_post,
                                                          (float*)d_out, N, H);
}
